// DependencyBertSelfAttetion_54056458387969
// MI455X (gfx1250) — matrix operand order repaired and checked by reading, not compiled, not run on hardware
//
#include <hip/hip_runtime.h>
#include <math.h>

typedef __bf16 bf16;
typedef __attribute__((ext_vector_type(16))) __bf16 v16bf;
typedef __attribute__((ext_vector_type(8)))  float  v8f;

#define B_ 8
#define T_ 1024
#define C_ 768
#define H_ 12
#define D_ 64

// ---------- WMMA fragment helpers (layouts per CDNA5 ISA 7.12.2) ----------

// A fragment 16x32 (MxK), source row-major [m][k] with row stride ld.
static __device__ __forceinline__ v16bf load_a_bf16(const bf16* __restrict__ base,
                                                    int ld, int lane) {
  int m = lane & 15, hi = lane >> 4;
  v16bf a;
#pragma unroll
  for (int i = 0; i < 8; ++i) {
    int k = ((i & 4) ? 16 : 0) + ((i & 3) * 2) + hi * 8;
    a[2 * i]     = base[m * ld + k];
    a[2 * i + 1] = base[m * ld + k + 1];
  }
  return a;
}

// B fragment 32x16 (KxN) where memory holds [n][k] row-major (k contiguous:
// lane-pairs merge into 32-bit loads). Used for Q*K^T, bf16 weights, V^T.
static __device__ __forceinline__ v16bf load_b_ncontig(const bf16* __restrict__ base,
                                                       int ld, int lane) {
  int n = lane & 15, kb = (lane >> 4) * 8;
  v16bf b;
#pragma unroll
  for (int i = 0; i < 8; ++i) {
    int k = ((i & 4) ? 16 : 0) + ((i & 3) * 2) + kb;
    b[2 * i]     = base[n * ld + k];
    b[2 * i + 1] = base[n * ld + k + 1];
  }
  return b;
}

static __device__ __forceinline__ v8f wmma_bf16(v16bf a, v16bf b, v8f c) {
  return __builtin_amdgcn_wmma_f32_16x16x32_bf16(false, a, false, b,
                                                 (short)0, c, false, false);
}

// ---------- kernel 1: f32 -> bf16 convert ----------
__global__ void cvt_bf16_kernel(const float* __restrict__ x, bf16* __restrict__ y, int n) {
  int i = blockIdx.x * blockDim.x + threadIdx.x;
  int stride = gridDim.x * blockDim.x;
  for (; i < n; i += stride) y[i] = (bf16)x[i];
}

// ---------- kernel 2: QKV projection GEMM ----------
// One wave per block computes a 16(M) x 64(N) strip: A fragment loaded once per
// K-step, reused by 4 WMMAs. Weights already bf16 (pre-converted, L2-resident).
// TRANSPOSED=false: dst[b,h,t,d] (Q,K).  TRANSPOSED=true: dst[b,h,d,t] (V).
template <bool TRANSPOSED>
__global__ void __launch_bounds__(32) qkv_gemm_kernel(const bf16* __restrict__ X,
                                const bf16* __restrict__ Wb,
                                const float* __restrict__ bias,
                                bf16* __restrict__ dst) {
  int lane = threadIdx.x & 31;
  int mbase = blockIdx.x * 16;
  int nbase = blockIdx.y * 64;
  v8f zero = {};
  v8f acc[4] = {zero, zero, zero, zero};
  for (int kb = 0; kb < C_; kb += 32) {
    v16bf a = load_a_bf16(X + (size_t)mbase * C_ + kb, C_, lane);
#pragma unroll
    for (int j = 0; j < 4; ++j) {
      v16bf b = load_b_ncontig(Wb + (size_t)(nbase + j * 16) * C_ + kb, C_, lane);
      acc[j] = wmma_bf16(a, b, acc[j]);
    }
  }
  int hi = lane >> 4;
  int bidx = mbase / T_;           // 16-row tile never crosses a batch boundary
  int tbase = mbase % T_;
#pragma unroll
  for (int j = 0; j < 4; ++j) {
    int n = nbase + j * 16 + (lane & 15);
    float bn = bias[n];
    int h = n / D_, d = n % D_;
#pragma unroll
    for (int r = 0; r < 8; ++r) {
      int t = tbase + (hi ? r + 8 : r);
      float val = acc[j][r] + bn;
      if (TRANSPOSED)
        dst[(((size_t)bidx * H_ + h) * D_ + d) * T_ + t] = (bf16)val;
      else
        dst[(((size_t)bidx * H_ + h) * T_ + t) * D_ + d] = (bf16)val;
    }
  }
}

// ---------- kernel 3: dual-branch flash attention ----------
// grid = (T/16, B*H), block = 32 (one wave per 16-row query tile)
// V is stored transposed [B,H,D,T] so PV B-fragments are k-contiguous.
// launch_bounds(32,1): relax VGPR budget so nothing spills (plenty of blocks
// in flight to cover latency at low waves/SIMD).
__global__ void __launch_bounds__(32, 1)
dep_attn_kernel(const bf16* __restrict__ Qb,
                const bf16* __restrict__ Kb,
                const bf16* __restrict__ Vt,
                const float* __restrict__ dep,
                float* __restrict__ osa,
                float* __restrict__ dsa) {
  __shared__ bf16 Pl[2][16][32];   // prob tiles, C-layout -> A-layout staging
  int lane = threadIdx.x & 31;
  int hi = lane >> 4, nlo = lane & 15;
  int bh = blockIdx.y;
  int b = bh / H_, h = bh % H_;
  int qbase = blockIdx.x * 16;

  const bf16* Qh = Qb + (size_t)bh * T_ * D_;
  const bf16* Kh = Kb + (size_t)bh * T_ * D_;
  const bf16* Vh = Vt + (size_t)bh * D_ * T_;   // [d][t]
  const float* depb = dep + (size_t)b * T_ * T_;

  v16bf qa0 = load_a_bf16(Qh + (size_t)qbase * D_ + 0,  D_, lane);
  v16bf qa1 = load_a_bf16(Qh + (size_t)qbase * D_ + 32, D_, lane);

  v8f zero = {};
  v8f o[2][4];
  float mrow[2][8];    // running row max (uniform across a row's 16 lanes)
  float lsum[2][8];    // PER-LANE partial row sum; reduced once at the end
#pragma unroll
  for (int br = 0; br < 2; ++br) {
#pragma unroll
    for (int c = 0; c < 4; ++c) o[br][c] = zero;
#pragma unroll
    for (int r = 0; r < 8; ++r) { mrow[br][r] = -1e30f; lsum[br][r] = 0.f; }
  }

  const float scale = 0.125f;  // 1/sqrt(64)

  for (int sb = 0; sb < T_; sb += 32) {
    // scores: two 16x16 tiles, K-dim = D = 64 (2 WMMAs each)
    v8f s0 = zero, s1 = zero;
    {
      v16bf b0 = load_b_ncontig(Kh + (size_t)(sb + 0)  * D_ + 0,  D_, lane);
      s0 = wmma_bf16(qa0, b0, s0);
      v16bf b1 = load_b_ncontig(Kh + (size_t)(sb + 0)  * D_ + 32, D_, lane);
      s0 = wmma_bf16(qa1, b1, s0);
      v16bf b2 = load_b_ncontig(Kh + (size_t)(sb + 16) * D_ + 0,  D_, lane);
      s1 = wmma_bf16(qa0, b2, s1);
      v16bf b3 = load_b_ncontig(Kh + (size_t)(sb + 16) * D_ + 32, D_, lane);
      s1 = wmma_bf16(qa1, b3, s1);
    }
#pragma unroll
    for (int br = 0; br < 2; ++br) {
      float x0[8], x1[8], bm[8];
#pragma unroll
      for (int r = 0; r < 8; ++r) {
        int row = hi ? r + 8 : r;
        int t = qbase + row;
        x0[r] = s0[r] * scale;
        x1[r] = s1[r] * scale;
        if (br == 1) {
          x0[r] *= depb[(size_t)t * T_ + sb + nlo];
          x1[r] *= depb[(size_t)t * T_ + sb + 16 + nlo];
        }
        bm[r] = fmaxf(x0[r], x1[r]);
      }
      // batched butterfly max across the 16 lanes of each row:
      // 8 independent permutes per step -> they pipeline under one wait
#pragma unroll
      for (int off = 1; off < 16; off <<= 1) {
        float t8[8];
#pragma unroll
        for (int r = 0; r < 8; ++r) t8[r] = __shfl_xor(bm[r], off, 32);
#pragma unroll
        for (int r = 0; r < 8; ++r) bm[r] = fmaxf(bm[r], t8[r]);
      }
#pragma unroll
      for (int r = 0; r < 8; ++r) {
        int row = hi ? r + 8 : r;
        float mnew = fmaxf(mrow[br][r], bm[r]);
        float alpha = __expf(mrow[br][r] - mnew);
        mrow[br][r] = mnew;
        float p0 = __expf(x0[r] - mnew);
        float p1 = __expf(x1[r] - mnew);
        lsum[br][r] = lsum[br][r] * alpha + (p0 + p1);  // per-lane partial
#pragma unroll
        for (int c = 0; c < 4; ++c) o[br][c][r] *= alpha;
        Pl[br][row][nlo]      = (bf16)p0;
        Pl[br][row][nlo + 16] = (bf16)p1;
      }
    }
    __syncthreads();
    // P @ V : A from LDS (16x32); V^T B-fragments are k-contiguous 32-bit loads
#pragma unroll
    for (int br = 0; br < 2; ++br) {
      v16bf pa = load_a_bf16(&Pl[br][0][0], 32, lane);
#pragma unroll
      for (int c = 0; c < 4; ++c) {
        v16bf vb = load_b_ncontig(Vh + (size_t)(c * 16) * T_ + sb, T_, lane);
        o[br][c] = wmma_bf16(pa, vb, o[br][c]);
      }
    }
    __syncthreads();
  }

  // final row-sum reduction (batched butterflies, once per kernel not per block)
#pragma unroll
  for (int br = 0; br < 2; ++br) {
#pragma unroll
    for (int off = 1; off < 16; off <<= 1) {
      float t8[8];
#pragma unroll
      for (int r = 0; r < 8; ++r) t8[r] = __shfl_xor(lsum[br][r], off, 32);
#pragma unroll
      for (int r = 0; r < 8; ++r) lsum[br][r] += t8[r];
    }
  }

#pragma unroll
  for (int r = 0; r < 8; ++r) {
    int t = qbase + (hi ? r + 8 : r);
    size_t rowoff = ((size_t)b * T_ + t) * C_ + (size_t)h * D_;
    float inv1 = 1.0f / lsum[0][r];
    float inv2 = 1.0f / lsum[1][r];
#pragma unroll
    for (int c = 0; c < 4; ++c) {
      osa[rowoff + c * 16 + nlo] = o[0][c][r] * inv1;
      dsa[rowoff + c * 16 + nlo] = o[1][c][r] * inv2;
    }
  }
}

// ---------- kernel 4: gated fusion ----------
// one block per token; g = sigmoid(tanh([osa,dsa]) . Wg + bg)
__global__ void gate_fuse_kernel(const float* __restrict__ osa,
                                 const float* __restrict__ dsa,
                                 const float* __restrict__ Wg,
                                 const float* __restrict__ bg,
                                 float* __restrict__ out) {
  __shared__ float red[8];
  __shared__ float gsh;
  int tok = blockIdx.x;
  const float* po = osa + (size_t)tok * C_;
  const float* pd = dsa + (size_t)tok * C_;
  float partial = 0.f;
  for (int c = threadIdx.x; c < C_; c += blockDim.x)
    partial += tanhf(po[c]) * Wg[c] + tanhf(pd[c]) * Wg[C_ + c];
#pragma unroll
  for (int off = 16; off; off >>= 1) partial += __shfl_xor(partial, off, 32);
  if ((threadIdx.x & 31) == 0) red[threadIdx.x >> 5] = partial;
  __syncthreads();
  if (threadIdx.x == 0) {
    float s = 0.f;
    for (int w = 0; w < (int)(blockDim.x >> 5); ++w) s += red[w];
    gsh = 1.0f / (1.0f + __expf(-(s + bg[0])));
  }
  __syncthreads();
  float g = gsh;
  float* pout = out + (size_t)tok * C_;
  for (int c = threadIdx.x; c < C_; c += blockDim.x)
    pout[c] = g * po[c] + (1.f - g) * pd[c];
}

extern "C" void kernel_launch(void* const* d_in, const int* in_sizes, int n_in,
                              void* d_out, int out_size, void* d_ws, size_t ws_size,
                              hipStream_t stream) {
  (void)in_sizes; (void)n_in; (void)out_size; (void)ws_size;
  const float* hs  = (const float*)d_in[0];
  const float* dep = (const float*)d_in[1];
  const float* Wq  = (const float*)d_in[2];
  const float* bq  = (const float*)d_in[3];
  const float* Wk  = (const float*)d_in[4];
  const float* bk  = (const float*)d_in[5];
  const float* Wv  = (const float*)d_in[6];
  const float* bv  = (const float*)d_in[7];
  const float* Wg  = (const float*)d_in[8];
  const float* bg  = (const float*)d_in[9];
  float* out = (float*)d_out;

  const size_t nTok = (size_t)B_ * T_;   // 8192
  const size_t nX   = nTok * C_;         // 6.29M elements
  const size_t nW   = (size_t)C_ * C_;   // 589824 elements

  char* ws = (char*)d_ws;
  bf16* hsb = (bf16*)ws;  ws += nX * sizeof(bf16);
  bf16* qb  = (bf16*)ws;  ws += nX * sizeof(bf16);
  bf16* kb  = (bf16*)ws;  ws += nX * sizeof(bf16);
  bf16* vb  = (bf16*)ws;  ws += nX * sizeof(bf16);   // stored transposed [B,H,D,T]
  bf16* wqb = (bf16*)ws;  ws += nW * sizeof(bf16);
  bf16* wkb = (bf16*)ws;  ws += nW * sizeof(bf16);
  bf16* wvb = (bf16*)ws;  ws += nW * sizeof(bf16);
  float* osa = (float*)ws; ws += nX * sizeof(float);
  float* dsa = (float*)ws; ws += nX * sizeof(float);
  // total workspace: ~104 MB

  cvt_bf16_kernel<<<1024, 256, 0, stream>>>(hs, hsb, (int)nX);
  cvt_bf16_kernel<<<576, 256, 0, stream>>>(Wq, wqb, (int)nW);
  cvt_bf16_kernel<<<576, 256, 0, stream>>>(Wk, wkb, (int)nW);
  cvt_bf16_kernel<<<576, 256, 0, stream>>>(Wv, wvb, (int)nW);

  dim3 ggrid((unsigned)(nTok / 16), C_ / 64);
  qkv_gemm_kernel<false><<<ggrid, 32, 0, stream>>>(hsb, wqb, bq, qb);
  qkv_gemm_kernel<false><<<ggrid, 32, 0, stream>>>(hsb, wkb, bk, kb);
  qkv_gemm_kernel<true ><<<ggrid, 32, 0, stream>>>(hsb, wvb, bv, vb);

  dim3 agrid(T_ / 16, B_ * H_);
  dep_attn_kernel<<<agrid, 32, 0, stream>>>(qb, kb, vb, dep, osa, dsa);

  gate_fuse_kernel<<<(unsigned)nTok, 256, 0, stream>>>(osa, dsa, Wg, bg, out);
}
